// GNN_node_45621142618640
// MI455X (gfx1250) — compile-verified, matrix operand order checked by reading, not run on hardware
//
#include <hip/hip_runtime.h>
#include <hip/hip_bf16.h>

// ---------------------------------------------------------------------------
// GNN pipeline for MI455X (gfx1250, wave32, WMMA).
//   - diffusion:   V_WMMA_F32_16X16X4_F32 (f32 GEMM, per-graph, all-LDS)
//   - GIN MLPs:    V_WMMA_F32_16X16X32_BF16 (bf16 in, f32 accumulate)
//   - msg passing: per-graph LDS float atomics (ds_add_f32)
//   - tile staging: GLOBAL_LOAD_ASYNC_TO_LDS_B128 (ASYNCcnt + s_wait_asynccnt)
// Workspace use: ~277 MB.
// ---------------------------------------------------------------------------

typedef __attribute__((ext_vector_type(16))) __bf16 v16bf;
typedef __attribute__((ext_vector_type(8)))  __bf16 v8bf;
typedef __attribute__((ext_vector_type(8)))  float  v8f;
typedef __attribute__((ext_vector_type(2)))  float  v2f;

#define B_  512
#define N_  128
#define E_  512
#define D_  300
#define FA_ 9
#define FB_ 3
#define V_  64
#define KP1 320   // padded K for GEMM1 (300 -> 320)
#define NP1 640   // padded N for GEMM1 (600 -> 640)
#define KP2 640   // padded K for GEMM2 (600 -> 640)
#define NP2 384   // padded N rows for w2T (300 -> 3*128)

// ---- gfx1250 async global->LDS path (guarded; sync fallback otherwise) ----
#if defined(__has_builtin)
#  if __has_builtin(__builtin_amdgcn_global_load_async_to_lds_b128)
#    define HAVE_ASYNC_LDS 1
#  endif
#  if __has_builtin(__builtin_amdgcn_s_wait_asynccnt)
#    define HAVE_WAIT_ASYNC_BUILTIN 1
#  endif
#endif

#ifdef HAVE_ASYNC_LDS
// Builtin signature (from clang diagnostic): param 0 is AS(1) int4*, so use the
// exact gcc-vector int4 type; AS(1) prints as __device__ in HIP mode.
typedef int v4i_t __attribute__((vector_size(16)));
typedef __attribute__((address_space(1))) v4i_t* g_v4i_p;
typedef __attribute__((address_space(3))) v4i_t* l_v4i_p;
__device__ __forceinline__ void async_b128(const void* g, void* l) {
  __builtin_amdgcn_global_load_async_to_lds_b128((g_v4i_p)g, (l_v4i_p)l, 0, 0);
}
__device__ __forceinline__ void wait_async() {
#ifdef HAVE_WAIT_ASYNC_BUILTIN
  __builtin_amdgcn_s_wait_asynccnt(0);
#else
  asm volatile("s_wait_asynccnt 0x0" ::: "memory");
#endif
}
#endif

__device__ __forceinline__ __bf16 f2bf(float f) {
  unsigned u = __builtin_bit_cast(unsigned, f);
  unsigned r = u + 0x7FFFu + ((u >> 16) & 1u);   // round-to-nearest-even
  unsigned short h = (unsigned short)(r >> 16);
  return __builtin_bit_cast(__bf16, h);
}

// ---------------- small prep kernels ----------------------------------------

// tabw[f][v] = dot(bond_tab_root[f][v][:], edge_lin_w)
__global__ void k_tabw(const float* __restrict__ tab, const float* __restrict__ elw,
                       float* __restrict__ tabw) {
  int i = blockIdx.x * blockDim.x + threadIdx.x;
  if (i >= FB_ * V_) return;
  const float* row = tab + (size_t)i * D_;
  float s = 0.f;
  for (int d = 0; d < D_; ++d) s += row[d] * elw[d];
  tabw[i] = s;
}

// ew[b*E+e] = sigmoid(sum_f tabw[f][attr] + b)
__global__ void k_ew(const int* __restrict__ ea, const float* __restrict__ tabw,
                     const float* __restrict__ elb, float* __restrict__ ew) {
  int i = blockIdx.x * blockDim.x + threadIdx.x;
  if (i >= B_ * E_) return;
  float s = *elb;
  for (int f = 0; f < FB_; ++f) s += tabw[f * V_ + ea[i * FB_ + f]];
  ew[i] = 1.f / (1.f + __expf(-s));
}

// h0 = 0.8 * sum_f atom_tab[f][idx][d]
__global__ void k_encode(const int* __restrict__ xa, const float* __restrict__ tab,
                         float* __restrict__ h) {
  int i = blockIdx.x * blockDim.x + threadIdx.x;
  if (i >= B_ * N_ * D_) return;
  int node = i / D_, d = i - node * D_;
  float s = 0.f;
  for (int f = 0; f < FA_; ++f)
    s += tab[((size_t)f * V_ + xa[node * FA_ + f]) * D_ + d];
  h[i] = 0.8f * s;
}

__global__ void k_zero(float* __restrict__ p, int n) {
  int i = blockIdx.x * blockDim.x + threadIdx.x;
  if (i < n) p[i] = 0.f;
}

// adj[r][c] = adj[c][r] = w  (deterministic under dup edges via bitwise max,
// valid since sigmoid output > 0)
__global__ void k_scatter(const int* __restrict__ ei, const float* __restrict__ ew,
                          float* __restrict__ A) {
  int i = blockIdx.x * blockDim.x + threadIdx.x;
  if (i >= B_ * E_) return;
  int b = i / E_, e = i - b * E_;
  int r = ei[b * 2 * E_ + e];
  int c = ei[b * 2 * E_ + E_ + e];
  unsigned wbits = __float_as_uint(ew[i]);
  unsigned* Ab = (unsigned*)(A + (size_t)b * N_ * N_);
  atomicMax(&Ab[r * N_ + c], wbits);
  atomicMax(&Ab[c * N_ + r], wbits);
}

// symmetric normalization: A[i,j] = colsum(i)^-.5 * (adj+I)[i,j] * rowsum(j)^-.5
__global__ void __launch_bounds__(256) k_norm(float* __restrict__ A) {
  __shared__ float rs[N_], cs[N_];
  int b = blockIdx.x, tid = threadIdx.x;
  float* Ab = A + (size_t)b * N_ * N_;
  if (tid < N_) {
    float s = 1.f;                       // +1 for identity on the diagonal
    for (int j = 0; j < N_; ++j) s += Ab[tid * N_ + j];
    rs[tid] = s;                         // row sums  (adj.sum(1))
  } else {
    int i = tid - N_;
    float s = 1.f;
    for (int k = 0; k < N_; ++k) s += Ab[k * N_ + i];
    cs[i] = s;                           // col sums  (adj.sum(0))
  }
  __syncthreads();
  for (int idx = tid; idx < N_ * N_; idx += 256) {
    int i = idx >> 7, j = idx & 127;
    float v = Ab[idx] + (i == j ? 1.f : 0.f);
    Ab[idx] = rsqrtf(cs[i]) * v * rsqrtf(rs[j]);
  }
}

// ---------------- diffusion: y = (x + Ax + ... + A^ord x)/(ord+1) -----------
// One workgroup per graph. A (64KB f32) + x ping-pong + y in LDS = 160KB.
// f32 WMMA 16x16x4: per wave one 16-row M-tile, 4 N-tiles over 64-col chunk.
__global__ void __launch_bounds__(256, 1)
k_diffuse(const float* __restrict__ Aadj, float* __restrict__ h,
          const int* __restrict__ order_p) {
  __shared__ __attribute__((aligned(16))) float Ald[N_ * N_];     // 64 KB
  __shared__ __attribute__((aligned(16))) float xb[2][N_ * 64];   // 64 KB
  __shared__ __attribute__((aligned(16))) float yb[N_ * 64];      // 32 KB
  int b = blockIdx.x, tid = threadIdx.x;
  int ord = *order_p;
  const float* Ag = Aadj + (size_t)b * N_ * N_;
  float* hg = h + (size_t)b * N_ * D_;
#ifdef HAVE_ASYNC_LDS
  for (int i = tid * 4; i < N_ * N_; i += 256 * 4)   // 16B per async op
    async_b128(Ag + i, Ald + i);
  wait_async();
#else
  for (int i = tid; i < N_ * N_; i += 256) Ald[i] = Ag[i];
#endif
  int lane = tid & 31, wid = tid >> 5;
  int lr = lane & 15, kh = lane >> 4;
  const v8f vz = {};
  for (int c = 0; c < 5; ++c) {                 // 5 chunks of 64 cols (300->320)
    int cbase = c * 64;
    __syncthreads();
    for (int idx = tid; idx < N_ * 64; idx += 256) {
      int row = idx >> 6, col = idx & 63;
      int g = cbase + col;
      float v = (g < D_) ? hg[row * D_ + g] : 0.f;
      xb[0][idx] = v; yb[idx] = v;
    }
    int cur = 0;
    for (int it = 0; it < ord; ++it) {
      __syncthreads();
      v8f acc[4];
      acc[0] = vz; acc[1] = vz; acc[2] = vz; acc[3] = vz;
      int arow = wid * 16 + lr;
      for (int k = 0; k < N_; k += 4) {
        v2f a;
        a.x = Ald[arow * N_ + k + kh * 2];
        a.y = Ald[arow * N_ + k + kh * 2 + 1];
#pragma unroll
        for (int nt = 0; nt < 4; ++nt) {
          v2f bb;
          int col = nt * 16 + lr;
          bb.x = xb[cur][(k + kh * 2) * 64 + col];
          bb.y = xb[cur][(k + kh * 2 + 1) * 64 + col];
          acc[nt] = __builtin_amdgcn_wmma_f32_16x16x4_f32(
              false, a, false, bb, (short)0, acc[nt], false, false);
        }
      }
      int nxt = cur ^ 1;
#pragma unroll
      for (int nt = 0; nt < 4; ++nt)
        for (int v = 0; v < 8; ++v) {
          int m = wid * 16 + v + ((lane < 16) ? 0 : 8);
          int n = nt * 16 + lr;
          xb[nxt][m * 64 + n] = acc[nt][v];
        }
      __syncthreads();
      for (int idx = tid; idx < N_ * 64; idx += 256) yb[idx] += xb[nxt][idx];
      cur = nxt;
    }
    __syncthreads();
    float inv = 1.f / (float)(ord + 1);
    for (int idx = tid; idx < N_ * 64; idx += 256) {
      int row = idx >> 6, col = idx & 63;
      int g = cbase + col;
      if (g < D_) hg[row * D_ + g] = yb[idx] * inv;
    }
  }
}

// ---------------- GIN message passing (per-graph, LDS atomics) --------------
__global__ void __launch_bounds__(256, 1)
k_mp(const float* __restrict__ h, const int* __restrict__ ei,
     const int* __restrict__ ea, const float* __restrict__ btab,
     const float* __restrict__ eps, int layer, float* __restrict__ Z) {
  __shared__ float accS[N_ * D_];   // 150 KB
  int b = blockIdx.x, tid = threadIdx.x;
  for (int i = tid; i < N_ * D_; i += 256) accS[i] = 0.f;
  __syncthreads();
  const int* eib = ei + b * 2 * E_;
  const int* eab = ea + (size_t)b * E_ * FB_;
  const float* hb = h + (size_t)b * N_ * D_;
  for (int t = tid; t < E_ * D_; t += 256) {
    int e = t / D_, d = t - e * D_;
    int src = eib[e], dst = eib[E_ + e];
    float ee = btab[(0 * V_ + eab[e * FB_ + 0]) * D_ + d]
             + btab[(1 * V_ + eab[e * FB_ + 1]) * D_ + d]
             + btab[(2 * V_ + eab[e * FB_ + 2]) * D_ + d];
    float m = hb[src * D_ + d] + ee;
    if (m > 0.f) atomicAdd(&accS[dst * D_ + d], m);   // ds_add_f32
  }
  __syncthreads();
  float ep = 1.f + eps[layer];
  float* Zb = Z + (size_t)b * N_ * D_;
  for (int i = tid; i < N_ * D_; i += 256) Zb[i] = ep * hb[i] + accS[i];
}

// ---------------- per-layer weight prep: transpose+pad -> bf16, fold BN -----
__global__ void k_prep(const float* __restrict__ w1, const float* __restrict__ b1,
                       const float* __restrict__ g1, const float* __restrict__ bb1,
                       const float* __restrict__ w2, const float* __restrict__ b2,
                       const float* __restrict__ g2, const float* __restrict__ bb2,
                       int l, __bf16* __restrict__ w1T, __bf16* __restrict__ w2T,
                       float* __restrict__ G1, float* __restrict__ C1,
                       float* __restrict__ G2, float* __restrict__ C2) {
  int i = blockIdx.x * blockDim.x + threadIdx.x;
  const int n1 = NP1 * KP1;            // 204800
  const int n2 = NP2 * KP2;            // 245760
  if (i < n1) {
    int n = i / KP1, k = i - n * KP1;
    float v = (n < 600 && k < 300) ? w1[(size_t)l * 300 * 600 + k * 600 + n] : 0.f;
    w1T[i] = f2bf(v);
  } else if (i < n1 + n2) {
    int j = i - n1;
    int n = j / KP2, k = j - n * KP2;
    float v = (n < 300 && k < 600) ? w2[(size_t)l * 600 * 300 + k * 300 + n] : 0.f;
    w2T[j] = f2bf(v);
  } else if (i < n1 + n2 + NP1) {
    int n = i - n1 - n2;
    float g = 0.f, c = 0.f;
    if (n < 600) { g = g1[l * 600 + n]; c = g * b1[l * 600 + n] + bb1[l * 600 + n]; }
    G1[n] = g; C1[n] = c;
  } else if (i < n1 + n2 + NP1 + NP2) {
    int n = i - n1 - n2 - NP1;
    float g = 0.f, c = 0.f;
    if (n < 300) { g = g2[l * 300 + n]; c = g * b2[l * 300 + n] + bb2[l * 300 + n]; }
    G2[n] = g; C2[n] = c;
  }
}

// ---------------- bf16 WMMA GEMM: out = act(G[n]*(A@B)[m,n] + C[n]) ---------
// Block tile 128(M)x128(N)x32(K). 8 waves (4x2): each wave 2 M-tiles x 4 N-tiles.
union FragBF { v16bf v; v8bf h[2]; };

template <bool A_F32, bool OUT_BF16>
__global__ void __launch_bounds__(256)
k_gemm(const void* __restrict__ Ap, int lda, int kAct, int kPad,
       const __bf16* __restrict__ Bt, int ldb,
       const float* __restrict__ G, const float* __restrict__ C,
       void* __restrict__ Outp, int ldo, int nAct, int doRelu) {
  __shared__ __attribute__((aligned(16))) __bf16 As[128 * 32];   // 8 KB [m][k]
  __shared__ __attribute__((aligned(16))) __bf16 Bs[128 * 32];   // 8 KB [n][k]
  int tid = threadIdx.x;
  int m0 = blockIdx.x * 128, n0 = blockIdx.y * 128;
  int lane = tid & 31, wid = tid >> 5;
  int wm = wid & 3, wn = wid >> 2;
  int lr = lane & 15, kh = lane >> 4;
  const v8f vz = {};
  v8f acc[2][4];
#pragma unroll
  for (int mt = 0; mt < 2; ++mt)
    for (int nt = 0; nt < 4; ++nt) acc[mt][nt] = vz;

  int lrow = tid >> 1;               // 0..127
  int lhalf = (tid & 1) * 16;        // 0 or 16 (elements)

  for (int k0 = 0; k0 < kPad; k0 += 32) {
    // ---- stage A tile ----
    if constexpr (A_F32) {
      // f32 -> bf16 conversion must go through VGPRs
      const float* A = (const float*)Ap;
      const float* src = A + (size_t)(m0 + lrow) * lda + k0 + lhalf;
      __bf16* dst = As + lrow * 32 + lhalf;
      if (k0 + 32 <= kAct) {
        __builtin_prefetch(src + 32, 0, 0);   // next K-tile -> global_prefetch
#pragma unroll
        for (int j = 0; j < 16; j += 4) {
          float4 f = *(const float4*)(src + j);
          dst[j + 0] = f2bf(f.x); dst[j + 1] = f2bf(f.y);
          dst[j + 2] = f2bf(f.z); dst[j + 3] = f2bf(f.w);
        }
      } else {
#pragma unroll
        for (int j = 0; j < 16; ++j) {
          int k = k0 + lhalf + j;
          dst[j] = f2bf(k < kAct ? src[j] : 0.f);
        }
      }
    } else {
      const __bf16* A = (const __bf16*)Ap;
      const __bf16* src = A + (size_t)(m0 + lrow) * lda + k0 + lhalf;
      __bf16* dst = As + lrow * 32 + lhalf;
#ifdef HAVE_ASYNC_LDS
      async_b128(src, dst);
      async_b128(src + 8, dst + 8);
#else
      ((uint4*)dst)[0] = ((const uint4*)src)[0];
      ((uint4*)dst)[1] = ((const uint4*)src)[1];
#endif
    }
    // ---- stage B tile (already bf16, [n][k] contiguous) ----
    {
      const __bf16* src = Bt + (size_t)(n0 + lrow) * ldb + k0 + lhalf;
      __bf16* dst = Bs + lrow * 32 + lhalf;
#ifdef HAVE_ASYNC_LDS
      async_b128(src, dst);
      async_b128(src + 8, dst + 8);
#else
      ((uint4*)dst)[0] = ((const uint4*)src)[0];
      ((uint4*)dst)[1] = ((const uint4*)src)[1];
#endif
    }
#ifdef HAVE_ASYNC_LDS
    wait_async();
#endif
    __syncthreads();
    // ---- fragments per ISA 16-bit A layout: chunks at k = kh*8, 16+kh*8 ----
    FragBF af[2], bfm[4];
#pragma unroll
    for (int mt = 0; mt < 2; ++mt) {
      int r = wm * 32 + mt * 16 + lr;
      af[mt].h[0] = *(const v8bf*)(As + r * 32 + kh * 8);
      af[mt].h[1] = *(const v8bf*)(As + r * 32 + 16 + kh * 8);
    }
#pragma unroll
    for (int nt = 0; nt < 4; ++nt) {
      int n = wn * 64 + nt * 16 + lr;
      bfm[nt].h[0] = *(const v8bf*)(Bs + n * 32 + kh * 8);
      bfm[nt].h[1] = *(const v8bf*)(Bs + n * 32 + 16 + kh * 8);
    }
#pragma unroll
    for (int mt = 0; mt < 2; ++mt)
#pragma unroll
      for (int nt = 0; nt < 4; ++nt)
        acc[mt][nt] = __builtin_amdgcn_wmma_f32_16x16x32_bf16(
            false, af[mt].v, false, bfm[nt].v, (short)0, acc[mt][nt], false, false);
    __syncthreads();
  }
  // ---- epilogue: folded BN + optional relu ----
  int mbase = m0 + wm * 32 + ((lane < 16) ? 0 : 8);
#pragma unroll
  for (int mt = 0; mt < 2; ++mt)
    for (int nt = 0; nt < 4; ++nt)
      for (int v = 0; v < 8; ++v) {
        int m = mbase + mt * 16 + v;
        int n = n0 + wn * 64 + nt * 16 + lr;
        float o = G[n] * acc[mt][nt][v] + C[n];
        if (doRelu) o = fmaxf(o, 0.f);
        if constexpr (OUT_BF16) {
          ((__bf16*)Outp)[(size_t)m * ldo + n] = f2bf(o);
        } else {
          if (n < nAct) ((float*)Outp)[(size_t)m * ldo + n] = o;
        }
      }
}

// ---------------------------------------------------------------------------
extern "C" void kernel_launch(void* const* d_in, const int* in_sizes, int n_in,
                              void* d_out, int out_size, void* d_ws, size_t ws_size,
                              hipStream_t stream) {
  const int*   order_p    = (const int*)d_in[0];
  const int*   x_atom     = (const int*)d_in[1];
  const int*   edge_index = (const int*)d_in[2];
  const int*   edge_attr  = (const int*)d_in[3];
  const float* atom_tab   = (const float*)d_in[4];
  const float* bond_root  = (const float*)d_in[5];
  const float* edge_lin_w = (const float*)d_in[6];
  const float* edge_lin_b = (const float*)d_in[7];
  const float* bond_tabs  = (const float*)d_in[8];
  const float* eps        = (const float*)d_in[9];
  const float* w1         = (const float*)d_in[10];
  const float* b1         = (const float*)d_in[11];
  const float* bn1_g      = (const float*)d_in[12];
  const float* bn1_b      = (const float*)d_in[13];
  const float* w2         = (const float*)d_in[14];
  const float* b2         = (const float*)d_in[15];
  const float* bn_g       = (const float*)d_in[16];
  const float* bn_b       = (const float*)d_in[17];

  char* wsp = (char*)d_ws;
  auto alloc = [&](size_t bytes) {
    char* p = wsp; wsp += (bytes + 255) & ~(size_t)255; return p;
  };
  float*  Aadj = (float*)alloc((size_t)B_ * N_ * N_ * 4);      // 33.6 MB
  float*  hA   = (float*)alloc((size_t)B_ * N_ * D_ * 4);      // 78.6 MB
  float*  Z    = (float*)alloc((size_t)B_ * N_ * D_ * 4);      // 78.6 MB
  __bf16* z1b  = (__bf16*)alloc((size_t)B_ * N_ * NP1 * 2);    // 83.9 MB
  __bf16* w1T  = (__bf16*)alloc((size_t)NP1 * KP1 * 2);
  __bf16* w2T  = (__bf16*)alloc((size_t)NP2 * KP2 * 2);
  float*  G1   = (float*)alloc(NP1 * 4);
  float*  C1   = (float*)alloc(NP1 * 4);
  float*  G2   = (float*)alloc(NP2 * 4);
  float*  C2   = (float*)alloc(NP2 * 4);
  float*  ewb  = (float*)alloc((size_t)B_ * E_ * 4);
  float*  tabw = (float*)alloc(FB_ * V_ * 4);

  // --- front end: edge weights, node features, normalized adjacency ---
  k_tabw   <<<1, 256, 0, stream>>>(bond_root, edge_lin_w, tabw);
  k_ew     <<<(B_ * E_) / 256, 256, 0, stream>>>(edge_attr, tabw, edge_lin_b, ewb);
  k_encode <<<(B_ * N_ * D_) / 256, 256, 0, stream>>>(x_atom, atom_tab, hA);
  k_zero   <<<(B_ * N_ * N_) / 256, 256, 0, stream>>>(Aadj, B_ * N_ * N_);
  k_scatter<<<(B_ * E_) / 256, 256, 0, stream>>>(edge_index, ewb, Aadj);
  k_norm   <<<B_, 256, 0, stream>>>(Aadj);
  k_diffuse<<<B_, 256, 0, stream>>>(Aadj, hA, order_p);

  // --- 5 GIN layers: MP -> GEMM1(bf16 wmma) -> GEMM2(bf16 wmma) ---
  float* hin = hA;
  float* hout = (float*)d_out;
  for (int l = 0; l < 5; ++l) {
    k_prep<<<1764, 256, 0, stream>>>(w1, b1, bn1_g, bn1_b, w2, b2, bn_g, bn_b,
                                     l, w1T, w2T, G1, C1, G2, C2);
    k_mp<<<B_, 256, 0, stream>>>(hin, edge_index, edge_attr,
                                 bond_tabs + (size_t)l * FB_ * V_ * D_, eps, l, Z);
    dim3 gg1(512, NP1 / 128);   // (512, 5)
    dim3 gg2(512, NP2 / 128);   // (512, 3)
    k_gemm<true,  true ><<<gg1, 256, 0, stream>>>(Z, D_, D_, KP1, w1T, KP1,
                                                  G1, C1, z1b, NP1, NP1, 1);
    k_gemm<false, false><<<gg2, 256, 0, stream>>>(z1b, NP1, NP1, KP2, w2T, KP2,
                                                  G2, C2, hout, D_, D_,
                                                  (l < 4) ? 1 : 0);
    float* t = hin; hin = hout; hout = t;
  }
  // layers write: l0->d_out, l1->hA, l2->d_out, l3->hA, l4->d_out  (final OK)
}
